// GraphGRUMortalityModel_32427003084952
// MI455X (gfx1250) — matrix operand order repaired and checked by reading, not compiled, not run on hardware
//
#include <hip/hip_runtime.h>
#include <math.h>

// ---------------- model constants ----------------
#define BB      128
#define TT      18
#define IN_DIM  64
#define HID     128
#define HEADS   4
#define DHEAD   32
#define N_CORE  1024
#define NNODES  ((BB + N_CORE) * TT)      // 20736
#define EE      331776
#define BT      (BB * TT)                  // 2304
#define G3      (3 * HID)                  // 384
#define CLS_IN  768

typedef __attribute__((ext_vector_type(16))) _Float16 v16h;
typedef __attribute__((ext_vector_type(8)))  float    v8f;

// ---------------- WMMA GEMM ----------------
// C[M,N] (ldc) = act(A[M,K] @ W[K,N] + bias)
// Template: K, N compile-time (full unroll, immediate-offset strided loads),
// NT = 16-col tiles per wave (1 or 2).
// grid = (M/16, N/(64*NT)), block = 128 threads (4 waves).
// Each wave owns a 16 x (16*NT) output tile; A fragment shared across NT.
template<int K, int N, int NT>
__global__ void gemm_wmma(const float* __restrict__ A, const float* __restrict__ W,
                          const float* __restrict__ bias, float* __restrict__ C,
                          int ldc, int relu) {
    const int lane   = threadIdx.x & 31;
    const int wave   = threadIdx.x >> 5;
    const int m0     = blockIdx.x * 16;
    const int n0     = (blockIdx.y * 4 + wave) * (16 * NT);
    const int laneHi = lane >> 4;              // 0 or 1
    const int row    = m0 + (lane & 15);
    const int colLo  = lane & 15;
    const int kgA    = laneHi * 8;             // ISA 16-bit A 16x32 layout
    const int kgB    = laneHi * 8;             // ISA 16-bit B 32x16 layout

    v8f acc[NT];
    #pragma unroll
    for (int nt = 0; nt < NT; ++nt) acc[nt] = (v8f){};

    #pragma unroll
    for (int k0 = 0; k0 < K; k0 += 32) {
        // A fragment: two contiguous 8-float runs -> 4x global_load_b128
        const float* Arow = A + (size_t)row * K + k0 + kgA;
        float4 a0 = ((const float4*)Arow)[0];
        float4 a1 = ((const float4*)Arow)[1];          // K = kgA+0..7
        float4 a2 = ((const float4*)(Arow + 16))[0];
        float4 a3 = ((const float4*)(Arow + 16))[1];   // K = kgA+16..23
        float af[16] = {a0.x, a0.y, a0.z, a0.w, a1.x, a1.y, a1.z, a1.w,
                        a2.x, a2.y, a2.z, a2.w, a3.x, a3.y, a3.z, a3.w};
        v16h a;
        #pragma unroll
        for (int j = 0; j < 16; ++j) a[j] = (_Float16)af[j];

        #pragma unroll
        for (int nt = 0; nt < NT; ++nt) {
            // B fragment: stride-N rows; N compile-time -> immediate offsets
            const float* Wp = W + (size_t)(k0 + kgB) * N + n0 + nt * 16 + colLo;
            v16h b;
            #pragma unroll
            for (int j = 0; j < 16; ++j)
                b[j] = (_Float16)Wp[(j < 8 ? j : j + 8) * N];
            acc[nt] = __builtin_amdgcn_wmma_f32_16x16x32_f16(false, a, false, b,
                                                             (short)0, acc[nt],
                                                             false, false);
        }
    }

    #pragma unroll
    for (int nt = 0; nt < NT; ++nt) {
        const int col = n0 + nt * 16 + colLo;
        const float bb = bias ? bias[col] : 0.0f;
        #pragma unroll
        for (int r = 0; r < 8; ++r) {              // ISA f32 C/D 16x16 layout
            int orow  = m0 + r + laneHi * 8;
            float val = acc[nt][r] + bb;
            if (relu) val = fmaxf(val, 0.0f);
            C[(size_t)orow * ldc + col] = val;
        }
    }
}

// ---------------- elementwise / edge kernels ----------------
__global__ void fill_f32(float* p, float val, int n) {
    int i = blockIdx.x * blockDim.x + threadIdx.x;
    if (i < n) p[i] = val;
}

__device__ __forceinline__ void atomicMaxF(float* addr, float val) {
    if (val >= 0.0f) atomicMax((int*)addr, __float_as_int(val));
    else             atomicMin((unsigned int*)addr, __float_as_uint(val));
}

__global__ void edge_logit_max(const int* __restrict__ ei, const float* __restrict__ q,
                               const float* __restrict__ k, float* __restrict__ logit,
                               float* __restrict__ m) {
    int i = blockIdx.x * blockDim.x + threadIdx.x;
    if (i >= EE * HEADS) return;
    int e = i >> 2, hh = i & 3;
    int s = ei[e], d = ei[EE + e];
    const float* qp = q + (size_t)d * HID + hh * DHEAD;
    const float* kp = k + (size_t)s * HID + hh * DHEAD;
    float acc = 0.0f;
    #pragma unroll
    for (int j = 0; j < DHEAD; ++j) acc += qp[j] * kp[j];
    acc *= 0.17677669529663687f;  // 1/sqrt(32)
    logit[i] = acc;
    atomicMaxF(&m[d * HEADS + hh], acc);
}

__global__ void edge_expsum(const int* __restrict__ ei, float* __restrict__ lw,
                            const float* __restrict__ m, float* __restrict__ z) {
    int i = blockIdx.x * blockDim.x + threadIdx.x;
    if (i >= EE * HEADS) return;
    int e = i >> 2, hh = i & 3;
    int d = ei[EE + e];
    float w = expf(lw[i] - m[d * HEADS + hh]);
    lw[i] = w;
    atomicAdd(&z[d * HEADS + hh], w);
}

__global__ void edge_agg(const int* __restrict__ ei, const float* __restrict__ lw,
                         const float* __restrict__ v, float* __restrict__ agg) {
    int i = blockIdx.x * blockDim.x + threadIdx.x;
    if (i >= EE * HID) return;
    int e = i >> 7, d = i & 127, hh = d >> 5;
    int sN = ei[e], dN = ei[EE + e];
    atomicAdd(&agg[(size_t)dN * HID + d], lw[e * HEADS + hh] * v[(size_t)sN * HID + d]);
}

__global__ void gat_finalize(const float* __restrict__ agg, const float* __restrict__ z,
                             const float* __restrict__ s, float* __restrict__ h) {
    int i = blockIdx.x * blockDim.x + threadIdx.x;
    if (i >= NNODES * HID) return;
    int n = i >> 7, d = i & 127, hh = d >> 5;
    float val = agg[i] / (z[n * HEADS + hh] + 1e-16f) + s[i];
    h[i] = fmaxf(val, 0.0f);
}

// ---------------- GRU ----------------
__global__ void transpose_w(const float* __restrict__ W, float* __restrict__ WT,
                            int R, int Ccols) {   // W: RxC -> WT: CxR
    int i = blockIdx.x * blockDim.x + threadIdx.x;
    if (i >= R * Ccols) return;
    int r = i / Ccols, c = i % Ccols;
    WT[(size_t)c * R + r] = W[i];
}

__global__ void gru_gate(const float* __restrict__ gx, const float* __restrict__ gh,
                         float* __restrict__ hst, float* __restrict__ ys, int t) {
    int i = blockIdx.x * blockDim.x + threadIdx.x;
    if (i >= BB * HID) return;
    int b = i >> 7, d = i & 127;
    const float* gxp = gx + (size_t)(b * TT + t) * G3;
    const float* ghp = gh + (size_t)b * G3;
    float r  = 1.0f / (1.0f + expf(-(gxp[d]           + ghp[d])));
    float zt = 1.0f / (1.0f + expf(-(gxp[HID + d]     + ghp[HID + d])));
    float n  = tanhf(gxp[2 * HID + d] + r * ghp[2 * HID + d]);
    float hp = hst[i];
    float hn = (1.0f - zt) * n + zt * hp;
    hst[i] = hn;
    ys[(size_t)(b * TT + t) * HID + d] = hn;
}

// ---------------- pooling / features / heads ----------------
__global__ void pool_kernel(const float* __restrict__ ys, const float* __restrict__ pmask,
                            float* __restrict__ xc) {
    int i = blockIdx.x * blockDim.x + threadIdx.x;
    if (i >= BB * HID) return;
    int b = i >> 7, d = i & 127;
    float len = 0.0f;
    #pragma unroll
    for (int t = 0; t < TT; ++t) len += pmask[b * TT + t];
    int mi = (int)len - 1;
    float last = ys[(size_t)(b * TT + mi) * HID + d];
    float sum = 0.0f, mx = -INFINITY;
    #pragma unroll
    for (int t = 0; t < TT; ++t) {
        float val = ys[(size_t)(b * TT + t) * HID + d] * pmask[b * TT + t];
        sum += val;
        mx = fmaxf(mx, val);
    }
    xc[(size_t)b * CLS_IN + d]            = last;
    xc[(size_t)b * CLS_IN + HID + d]      = sum / len;
    xc[(size_t)b * CLS_IN + 2 * HID + d]  = mx;
}

__global__ void pres_kernel(const int* __restrict__ ids, const float* __restrict__ emb,
                            float* __restrict__ xc) {
    int i = blockIdx.x * blockDim.x + threadIdx.x;
    if (i >= BB * HID) return;
    int b = i >> 7, d = i & 127;
    float acc = 0.0f;
    #pragma unroll
    for (int j = 0; j < 20; ++j) acc += emb[(size_t)ids[b * 20 + j] * HID + d];
    xc[(size_t)b * CLS_IN + 5 * HID + d] = fmaxf(acc * (1.0f / 20.0f), 0.0f);
}

__global__ void head_final(const float* __restrict__ h2, const float* __restrict__ W3,
                           const float* __restrict__ b3, float* __restrict__ out, int col) {
    int b = blockIdx.x * blockDim.x + threadIdx.x;
    if (b >= BB) return;
    float acc = b3[0];
    #pragma unroll
    for (int j = 0; j < 64; ++j) acc += h2[b * 64 + j] * W3[j];
    out[b * 3 + col] = acc;
}

// ---------------- host orchestration ----------------
extern "C" void kernel_launch(void* const* d_in, const int* in_sizes, int n_in,
                              void* d_out, int out_size, void* d_ws, size_t ws_size,
                              hipStream_t stream) {
    (void)in_sizes; (void)n_in; (void)out_size; (void)ws_size;
    const float* x       = (const float*)d_in[0];
    const float* pmask   = (const float*)d_in[1];
    const int*   eidx    = (const int*)  d_in[2];
    const float* nots    = (const float*)d_in[3];
    const float* bios    = (const float*)d_in[4];
    const int*   presid  = (const int*)  d_in[5];
    const float* X_core  = (const float*)d_in[6];
    const float* W_notes = (const float*)d_in[39];
    const float* b_notes = (const float*)d_in[40];
    const float* W_bios  = (const float*)d_in[41];
    const float* b_bios  = (const float*)d_in[42];
    const float* presemb = (const float*)d_in[43];
    float* out = (float*)d_out;

    // ---- workspace carve (floats) ----
    float* ws  = (float*)d_ws;
    float* h   = ws;                       // NNODES*128
    float* q   = h   + (size_t)NNODES * 128;
    float* kb  = q   + (size_t)NNODES * 128;
    float* v   = kb  + (size_t)NNODES * 128;
    float* s   = v   + (size_t)NNODES * 128;
    float* lw  = s   + (size_t)NNODES * 128;   // EE*4 (logit, then w)
    float* m   = lw  + (size_t)EE * 4;         // NNODES*4
    float* z   = m   + (size_t)NNODES * 4;     // NNODES*4
    float* agg = z   + (size_t)NNODES * 4;     // NNODES*128
    float* WT1 = agg + (size_t)NNODES * 128;   // 128*384
    float* WT2 = WT1 + 128 * G3;
    float* gx  = WT2 + 128 * G3;               // 2304*384
    float* gh  = gx  + (size_t)BT * G3;        // 128*384
    float* hst = gh  + BB * G3;                // 128*128
    float* ysA = hst + BB * HID;               // 2304*128
    float* ysB = ysA + (size_t)BT * HID;
    float* xc  = ysB + (size_t)BT * HID;       // 128*768
    float* mh1 = xc  + BB * CLS_IN;            // 128*128
    float* mh2 = mh1 + BB * HID;               // 128*64

    // dispatch over compile-time (K, N); NT=2 for N in {128,384}, NT=1 for N=64
    auto gemm = [&](const float* A, const float* W, const float* bias, float* C,
                    int M, int K, int N, int ldc, int relu) {
        if (N == 128) {
            dim3 g(M / 16, 1);
            switch (K) {
            case 64:  gemm_wmma<64, 128,2><<<g, dim3(128), 0, stream>>>(A, W, bias, C, ldc, relu); break;
            case 128: gemm_wmma<128,128,2><<<g, dim3(128), 0, stream>>>(A, W, bias, C, ldc, relu); break;
            case 768: gemm_wmma<768,128,2><<<g, dim3(128), 0, stream>>>(A, W, bias, C, ldc, relu); break;
            }
        } else if (N == 384) {   // K == 128 (GRU)
            dim3 g(M / 16, 3);
            gemm_wmma<128, 384, 2><<<g, dim3(128), 0, stream>>>(A, W, bias, C, ldc, relu);
        } else {                 // N == 64, K == 128 (MLP mid layer)
            dim3 g(M / 16, 1);
            gemm_wmma<128, 64, 1><<<g, dim3(128), 0, stream>>>(A, W, bias, C, ldc, relu);
        }
    };
    auto fill = [&](float* p, float val, int n) {
        fill_f32<<<(n + 255) / 256, 256, 0, stream>>>(p, val, n);
    };

    // ================= GAT layers =================
    for (int l = 0; l < 3; ++l) {
        const float* Wq = (const float*)d_in[7 + l * 8 + 0];
        const float* bq = (const float*)d_in[7 + l * 8 + 1];
        const float* Wk = (const float*)d_in[7 + l * 8 + 2];
        const float* bk = (const float*)d_in[7 + l * 8 + 3];
        const float* Wv = (const float*)d_in[7 + l * 8 + 4];
        const float* bv = (const float*)d_in[7 + l * 8 + 5];
        const float* Ws = (const float*)d_in[7 + l * 8 + 6];
        const float* bs = (const float*)d_in[7 + l * 8 + 7];

        if (l == 0) {   // input is [x; X_core] (K=64), launch each GEMM in two parts
            const float* Wmat[4] = {Wq, Wk, Wv, Ws};
            const float* bvec[4] = {bq, bk, bv, bs};
            float*       Cout[4] = {q, kb, v, s};
            for (int p = 0; p < 4; ++p) {
                gemm(x,      Wmat[p], bvec[p], Cout[p],                 BT,          IN_DIM, HID, HID, 0);
                gemm(X_core, Wmat[p], bvec[p], Cout[p] + (size_t)BT * HID,
                                                                        NNODES - BT, IN_DIM, HID, HID, 0);
            }
        } else {
            gemm(h, Wq, bq, q,  NNODES, HID, HID, HID, 0);
            gemm(h, Wk, bk, kb, NNODES, HID, HID, HID, 0);
            gemm(h, Wv, bv, v,  NNODES, HID, HID, HID, 0);
            gemm(h, Ws, bs, s,  NNODES, HID, HID, HID, 0);
        }

        fill(m,   -INFINITY, NNODES * HEADS);
        fill(z,   0.0f,      NNODES * HEADS);
        fill(agg, 0.0f,      NNODES * HID);

        int ne = EE * HEADS;
        edge_logit_max<<<(ne + 255) / 256, 256, 0, stream>>>(eidx, q, kb, lw, m);
        edge_expsum   <<<(ne + 255) / 256, 256, 0, stream>>>(eidx, lw, m, z);
        int na = EE * HID;
        edge_agg      <<<(na + 255) / 256, 256, 0, stream>>>(eidx, lw, v, agg);
        gat_finalize  <<<(NNODES * HID + 255) / 256, 256, 0, stream>>>(agg, z, s, h);
    }

    // ================= GRU layers =================
    for (int l = 0; l < 2; ++l) {
        const float* Wih = (const float*)d_in[31 + l * 4 + 0];  // (384,128)
        const float* Whh = (const float*)d_in[31 + l * 4 + 1];  // (384,128)
        const float* bih = (const float*)d_in[31 + l * 4 + 2];
        const float* bhh = (const float*)d_in[31 + l * 4 + 3];
        transpose_w<<<(G3 * HID + 255) / 256, 256, 0, stream>>>(Wih, WT1, G3, HID);
        transpose_w<<<(G3 * HID + 255) / 256, 256, 0, stream>>>(Whh, WT2, G3, HID);

        const float* gin  = (l == 0) ? h   : ysA;   // layer0 input = first 2304 rows of h
        float*       yout = (l == 0) ? ysA : ysB;

        gemm(gin, WT1, bih, gx, BT, HID, G3, G3, 0);
        fill(hst, 0.0f, BB * HID);
        for (int t = 0; t < TT; ++t) {
            gemm(hst, WT2, bhh, gh, BB, HID, G3, G3, 0);
            gru_gate<<<(BB * HID + 255) / 256, 256, 0, stream>>>(gx, gh, hst, yout, t);
        }
    }

    // ================= pooling + aux features into xc =================
    pool_kernel<<<(BB * HID + 255) / 256, 256, 0, stream>>>(ysB, pmask, xc);
    gemm(nots, W_notes, b_notes, xc + 3 * HID, BB, 768, HID, CLS_IN, 1);   // nh
    gemm(bios, W_bios,  b_bios,  xc + 4 * HID, BB, 64,  HID, CLS_IN, 1);   // bh
    pres_kernel<<<(BB * HID + 255) / 256, 256, 0, stream>>>(presid, presemb, xc);

    // ================= MLP heads (output column order: p0->0, p1->2, p2->1) ====
    const int colmap[3] = {0, 2, 1};
    for (int i = 0; i < 3; ++i) {
        const float* W1 = (const float*)d_in[44 + i * 6 + 0];
        const float* b1 = (const float*)d_in[44 + i * 6 + 1];
        const float* W2 = (const float*)d_in[44 + i * 6 + 2];
        const float* b2 = (const float*)d_in[44 + i * 6 + 3];
        const float* W3 = (const float*)d_in[44 + i * 6 + 4];
        const float* b3 = (const float*)d_in[44 + i * 6 + 5];
        gemm(xc,  W1, b1, mh1, BB, CLS_IN, HID, HID, 1);
        gemm(mh1, W2, b2, mh2, BB, HID,    64,  64,  1);
        head_final<<<1, 128, 0, stream>>>(mh2, W3, b3, out, colmap[i]);
    }
}
